// DGCNN_30855045055028
// MI455X (gfx1250) — matrix operand order repaired and checked by reading, not compiled, not run on hardware
//
#include <hip/hip_runtime.h>
#include <hip/hip_bf16.h>

// ---------------------------------------------------------------------------
// DGCNN forward on gfx1250 (MI455X). Compute-bound (~185 GFLOP, ~100MB traffic)
// -> all large GEMMs on v_wmma_f32_16x16x32_f16; branch-free b128 inner loops;
// 4 WMMA col-tiles per wave per A fragment. Batch-stat BN forces a multi-
// kernel pipeline; stats fused into the GEMM epilogue.
// ---------------------------------------------------------------------------

typedef __attribute__((ext_vector_type(16))) _Float16 v16h;
typedef __attribute__((ext_vector_type(8)))  _Float16 v8h;
typedef __attribute__((ext_vector_type(8)))  float    v8f;

#define EPSV 1e-5f

__device__ __forceinline__ float lrelu(float v){ return v > 0.f ? v : 0.2f*v; }

// order-preserving float<->uint encoding for atomicMax-based float max
__device__ __forceinline__ unsigned fenc(float f){
    unsigned u = __float_as_uint(f);
    return (u & 0x80000000u) ? ~u : (u | 0x80000000u);
}
__device__ __forceinline__ float fdec(unsigned u){
    unsigned v = (u & 0x80000000u) ? (u & 0x7FFFFFFFu) : ~u;
    return __uint_as_float(v);
}
#define ENC_NEG_INF 0x007FFFFFu   // fenc(-inf)

// ---------------------------------------------------------------------------
// weight conversion + transpose: W[Kd x N] f32 -> Wt[N x Kd] f16
// (transposed so B-fragments become contiguous 16-element K-runs)
// ---------------------------------------------------------------------------
__global__ void k_wt_f16(const float* __restrict__ w, _Float16* __restrict__ wt,
                         int Kd, int N){
    int i = blockIdx.x*256 + threadIdx.x;
    if (i < Kd*N){
        int k = i / N, n = i % N;
        wt[(size_t)n*Kd + k] = (_Float16)w[i];
    }
}

// ---------------------------------------------------------------------------
// input MLP: (32768 x 6) @ (6 x 64), accumulate BN stats (sum, sumsq per col)
// ---------------------------------------------------------------------------
__global__ __launch_bounds__(256)
void k_input_mlp(const float* __restrict__ x, const float* __restrict__ w,
                 float* __restrict__ h, float* __restrict__ stats){
    __shared__ float ssum[64], ssq[64];
    if (threadIdx.x < 64){ ssum[threadIdx.x] = 0.f; ssq[threadIdx.x] = 0.f; }
    __syncthreads();
    int r = blockIdx.x*256 + threadIdx.x;           // 32768 rows
    float xi[6];
    #pragma unroll
    for (int i = 0; i < 6; i++) xi[i] = x[(size_t)r*6 + i];
    for (int j = 0; j < 64; j++){
        float acc = 0.f;
        #pragma unroll
        for (int i = 0; i < 6; i++) acc += xi[i] * w[i*64 + j];
        h[(size_t)r*64 + j] = acc;
        atomicAdd(&ssum[j], acc);
        atomicAdd(&ssq[j], acc*acc);
    }
    __syncthreads();
    if (threadIdx.x < 64){
        atomicAdd(&stats[threadIdx.x],    ssum[threadIdx.x]);
        atomicAdd(&stats[64+threadIdx.x], ssq[threadIdx.x]);
    }
}

// stats(sum,sumsq) -> (scale, shift) per channel
__global__ void k_bn_finalize(const float* __restrict__ stats,
                              const float* __restrict__ g, const float* __restrict__ b,
                              float* __restrict__ ss, int C, float invCount){
    int c = blockIdx.x*blockDim.x + threadIdx.x;
    if (c >= C) return;
    float m = stats[c] * invCount;
    float v = stats[C + c] * invCount - m*m;
    float sc = g[c] * rsqrtf(v + EPSV);
    ss[c]     = sc;
    ss[C + c] = b[c] - m*sc;
}

__global__ void k_bn_apply(const float* __restrict__ h, const float* __restrict__ ss,
                           float* __restrict__ o, int C, int total){
    int i = blockIdx.x*blockDim.x + threadIdx.x;
    if (i >= total) return;
    int c = i % C;
    o[i] = lrelu(h[i]*ss[c] + ss[C + c]);
}

// ---------------------------------------------------------------------------
// per-batch kNN (64 points, C channels, K=16), exclude self
// ---------------------------------------------------------------------------
__global__ __launch_bounds__(64)
void k_knn(const float* __restrict__ x, int* __restrict__ idx, int C){
    __shared__ float xs[64*64];
    __shared__ float dmat[64*64];
    int b = blockIdx.x, t = threadIdx.x;
    const float* xb = x + (size_t)b*64*C;
    for (int m = 0; m < 64; m++) dmat[t*64 + m] = 0.f;
    for (int c0 = 0; c0 < C; c0 += 64){
        __syncthreads();
        #pragma unroll 4
        for (int cc = 0; cc < 64; cc += 4){
            float4 v = *(const float4*)(xb + (size_t)t*C + c0 + cc);
            *(float4*)(xs + t*64 + cc) = v;
        }
        __syncthreads();
        for (int m = 0; m < 64; m++){
            float a = 0.f;
            for (int cc = 0; cc < 64; cc++){
                float dv = xs[t*64 + cc] - xs[m*64 + cc];
                a += dv*dv;
            }
            dmat[t*64 + m] += a;
        }
    }
    dmat[t*64 + t] = 1e30f;                 // exclude self
    for (int kk = 0; kk < 16; kk++){
        float best = 1e29f; int bi = 0;
        for (int m = 0; m < 64; m++){
            float v = dmat[t*64 + m];
            if (v < best){ best = v; bi = m; }
        }
        dmat[t*64 + bi] = 1e30f;
        idx[((size_t)b*64 + t)*16 + kk] = bi;
    }
}

// ---------------------------------------------------------------------------
// WMMA GEMM helpers
// ---------------------------------------------------------------------------
__device__ __forceinline__ float4 ld4(const float* p){ return *(const float4*)p; }
__device__ __forceinline__ float4 bn4(float4 v, float4 sc, float4 sh){
    v.x = lrelu(v.x*sc.x + sh.x);
    v.y = lrelu(v.y*sc.y + sh.y);
    v.z = lrelu(v.z*sc.z + sh.z);
    v.w = lrelu(v.w*sc.w + sh.w);
    return v;
}
__device__ __forceinline__ float4 sub4(float4 a, float4 b){
    return make_float4(a.x-b.x, a.y-b.y, a.z-b.z, a.w-b.w);
}

// A fragment: one contiguous 16-float K-run at s (run start % 16 == 0)
//   AK 0: raw      AK 1: lrelu(scale*v+shift)      AK 3: p0[k]-p1[k]
template<int AK>
__device__ __forceinline__ v16h fetchA(const float* __restrict__ p0,
                                       const float* __restrict__ p1,
                                       const float* __restrict__ scp,
                                       const float* __restrict__ shp, int s){
    float4 r0, r1, r2, r3;
    if (AK == 3){
        r0 = sub4(ld4(p0+s),    ld4(p1+s));
        r1 = sub4(ld4(p0+s+4),  ld4(p1+s+4));
        r2 = sub4(ld4(p0+s+8),  ld4(p1+s+8));
        r3 = sub4(ld4(p0+s+12), ld4(p1+s+12));
    } else {
        r0 = ld4(p0+s); r1 = ld4(p0+s+4); r2 = ld4(p0+s+8); r3 = ld4(p0+s+12);
        if (AK == 1){
            r0 = bn4(r0, ld4(scp+s),    ld4(shp+s));
            r1 = bn4(r1, ld4(scp+s+4),  ld4(shp+s+4));
            r2 = bn4(r2, ld4(scp+s+8),  ld4(shp+s+8));
            r3 = bn4(r3, ld4(scp+s+12), ld4(shp+s+12));
        }
    }
    v16h a;
    a[0]=(_Float16)r0.x;  a[1]=(_Float16)r0.y;  a[2]=(_Float16)r0.z;  a[3]=(_Float16)r0.w;
    a[4]=(_Float16)r1.x;  a[5]=(_Float16)r1.y;  a[6]=(_Float16)r1.z;  a[7]=(_Float16)r1.w;
    a[8]=(_Float16)r2.x;  a[9]=(_Float16)r2.y;  a[10]=(_Float16)r2.z; a[11]=(_Float16)r2.w;
    a[12]=(_Float16)r3.x; a[13]=(_Float16)r3.y; a[14]=(_Float16)r3.z; a[15]=(_Float16)r3.w;
    return a;
}

__device__ __forceinline__ v16h ldB(const _Float16* __restrict__ p, int off){
    const v8h* q = (const v8h*)(p + off);
    v8h lo = q[0], hi = q[1];
    return __builtin_shufflevector(lo, hi, 0,1,2,3,4,5,6,7,8,9,10,11,12,13,14,15);
}

// 4 independent WMMAs off one A fragment (no D->A/B hazards between them)
__device__ __forceinline__ void wmma4(const v16h& a,
        const _Float16* __restrict__ pB0, const _Float16* __restrict__ pB1,
        const _Float16* __restrict__ pB2, const _Float16* __restrict__ pB3,
        int off, v8f& c0, v8f& c1, v8f& c2, v8f& c3){
    v16h b0 = ldB(pB0, off), b1 = ldB(pB1, off);
    v16h b2 = ldB(pB2, off), b3 = ldB(pB3, off);
    c0 = __builtin_amdgcn_wmma_f32_16x16x32_f16(false, a, false, b0, (short)0, c0, false, false);
    c1 = __builtin_amdgcn_wmma_f32_16x16x32_f16(false, a, false, b1, (short)0, c1, false, false);
    c2 = __builtin_amdgcn_wmma_f32_16x16x32_f16(false, a, false, b2, (short)0, c2, false, false);
    c3 = __builtin_amdgcn_wmma_f32_16x16x32_f16(false, a, false, b3, (short)0, c3, false, false);
}

// ---------------------------------------------------------------------------
// generic WMMA GEMM:  H[M x Nout] = A[M x Kdim] @ W[Kdim x Nout]
//   Wt is the f16 TRANSPOSED weight (Nout x Kdim) -> contiguous B fragments.
//   MODE 0: A = raw f32 buffer
//   MODE 1: A = lrelu(scale*h + shift) applied on load (BN fused)
//   MODE 2: A = edge features concat(x[nbr], x[ctr]-x[nbr]); since Cin%32==0
//           the K-loop splits into two uniform, branch-free halves.
// one wave per 16-row x 64-col tile (4 WMMA col-tiles); all fetches b128.
// ---------------------------------------------------------------------------
template<int MODE>
__global__ __launch_bounds__(128)
void k_gemm(const float* __restrict__ A, const float* __restrict__ ss,
            const int* __restrict__ idx, int Cin,
            const _Float16* __restrict__ Wt,      // Nout x Kdim (transposed)
            int M, int Kdim, int Nout,
            float* __restrict__ Hout, float* __restrict__ stats){
    __shared__ float ssum[64], ssq[64];
    if (threadIdx.x < 64){ ssum[threadIdx.x] = 0.f; ssq[threadIdx.x] = 0.f; }
    __syncthreads();

    int wave = threadIdx.x >> 5, lane = threadIdx.x & 31;
    int rowTile = blockIdx.x*4 + wave;
    int colQuad = blockIdx.y;                 // covers 64 output columns
    bool active = ((long)rowTile*16 < (long)M);

    int m    = lane & 15;        // A row within tile
    int hi   = lane >> 4;        // lane half selector
    int nloc = lane & 15;        // output column within a 16-col tile
    int n0   = colQuad*64 + nloc;

    const float *pA = nullptr, *pN = nullptr, *pC = nullptr;
    long row = (long)rowTile*16 + m;
    if (active){
        if (MODE == 2){
            int bb     = (int)(row >> 10);          // 64 pts * 16 nbrs
            int within = (int)(row & 1023);
            int ctr    = within >> 4;
            int nbr    = idx[row];
            pN = A + ((size_t)bb*64 + nbr)*Cin;
            pC = A + ((size_t)bb*64 + ctr)*Cin;
        } else {
            pA = A + (size_t)row*Kdim;
        }
    }
    const _Float16* pB0 = Wt + (size_t)(n0     )*Kdim;
    const _Float16* pB1 = Wt + (size_t)(n0 + 16)*Kdim;
    const _Float16* pB2 = Wt + (size_t)(n0 + 32)*Kdim;
    const _Float16* pB3 = Wt + (size_t)(n0 + 48)*Kdim;

    int offA = hi*16;       // lane's A-run base within a 32-wide K-step
    int offB = hi*16;       // lane's B-run base within a 32-wide K-step

    v8f c0 = {}, c1 = {}, c2 = {}, c3 = {};
    if (active){
        if (MODE == 2){
            // first half: A = x[nbr]          (uniform, branch-free)
            for (int kb = 0; kb < Cin; kb += 32){
                v16h a = fetchA<0>(pN, nullptr, nullptr, nullptr, kb + offA);
                wmma4(a, pB0, pB1, pB2, pB3, kb + offB, c0, c1, c2, c3);
            }
            // second half: A = x[ctr]-x[nbr]  (uniform, branch-free)
            for (int kb = Cin; kb < Kdim; kb += 32){
                v16h a = fetchA<3>(pC, pN, nullptr, nullptr, kb - Cin + offA);
                wmma4(a, pB0, pB1, pB2, pB3, kb + offB, c0, c1, c2, c3);
            }
        } else if (MODE == 1){
            for (int kb = 0; kb < Kdim; kb += 32){
                v16h a = fetchA<1>(pA, nullptr, ss, ss + Kdim, kb + offA);
                wmma4(a, pB0, pB1, pB2, pB3, kb + offB, c0, c1, c2, c3);
            }
        } else {
            for (int kb = 0; kb < Kdim; kb += 32){
                v16h a = fetchA<0>(pA, nullptr, nullptr, nullptr, kb + offA);
                wmma4(a, pB0, pB1, pB2, pB3, kb + offB, c0, c1, c2, c3);
            }
        }
        // epilogue: write + column stats. VGPR r: M=r (+8 upper lanes), N=nloc
        float s0=0.f,q0=0.f,s1=0.f,q1=0.f,s2=0.f,q2=0.f,s3=0.f,q3=0.f;
        #pragma unroll
        for (int r8 = 0; r8 < 8; r8++){
            int mm = r8 + hi*8;
            size_t ro = ((size_t)rowTile*16 + mm)*Nout + n0;
            float v0=c0[r8], v1=c1[r8], v2=c2[r8], v3=c3[r8];
            Hout[ro     ] = v0;
            Hout[ro + 16] = v1;
            Hout[ro + 32] = v2;
            Hout[ro + 48] = v3;
            s0+=v0; q0+=v0*v0; s1+=v1; q1+=v1*v1;
            s2+=v2; q2+=v2*v2; s3+=v3; q3+=v3*v3;
        }
        s0 += __shfl_xor(s0,16,32); q0 += __shfl_xor(q0,16,32);
        s1 += __shfl_xor(s1,16,32); q1 += __shfl_xor(q1,16,32);
        s2 += __shfl_xor(s2,16,32); q2 += __shfl_xor(q2,16,32);
        s3 += __shfl_xor(s3,16,32); q3 += __shfl_xor(q3,16,32);
        if (hi == 0){
            atomicAdd(&ssum[nloc     ], s0); atomicAdd(&ssq[nloc     ], q0);
            atomicAdd(&ssum[nloc + 16], s1); atomicAdd(&ssq[nloc + 16], q1);
            atomicAdd(&ssum[nloc + 32], s2); atomicAdd(&ssq[nloc + 32], q2);
            atomicAdd(&ssum[nloc + 48], s3); atomicAdd(&ssq[nloc + 48], q3);
        }
    }
    __syncthreads();
    if (threadIdx.x < 64){
        atomicAdd(&stats[colQuad*64 + threadIdx.x],        ssum[threadIdx.x]);
        atomicAdd(&stats[Nout + colQuad*64 + threadIdx.x], ssq[threadIdx.x]);
    }
}

// ---------------------------------------------------------------------------
// scatter-max per batch: out[idx[n,k]] = max over edges of lrelu(bn(h_edge))
// ---------------------------------------------------------------------------
__global__ __launch_bounds__(256)
void k_scatter_max(const float* __restrict__ H, const float* __restrict__ ss,
                   const int* __restrict__ idx, float* __restrict__ Xout, int C){
    __shared__ unsigned acc[64*128];
    int b = blockIdx.x;
    int chunks = (C + 127) / 128;
    for (int ch = 0; ch < chunks; ch++){
        int c0 = ch*128;
        int cw = (C - c0 < 128) ? (C - c0) : 128;
        for (int i = threadIdx.x; i < 64*cw; i += 256) acc[i] = ENC_NEG_INF;
        __syncthreads();
        for (int e = threadIdx.x; e < 1024; e += 256){
            long row = (long)b*1024 + e;
            int j = idx[row];
            const float* hr = H + (size_t)row*C;
            for (int c = 0; c < cw; c++){
                int cg = c0 + c;
                float v = lrelu(hr[cg]*ss[cg] + ss[C + cg]);
                atomicMax(&acc[j*cw + c], fenc(v));
            }
        }
        __syncthreads();
        for (int i = threadIdx.x; i < 64*cw; i += 256){
            int j = i / cw, c = i % cw;
            unsigned u = acc[i];
            float v = (u == ENC_NEG_INF) ? 0.f : fdec(u);
            Xout[((size_t)b*64 + j)*C + c0 + c] = v;
        }
        __syncthreads();
    }
}

// ---------------------------------------------------------------------------
// pooling: per batch, concat(mean over 64 pts, max over 64 pts) of [x1|x2|x3]
// ---------------------------------------------------------------------------
__global__ __launch_bounds__(256)
void k_pool(const float* __restrict__ x1, const float* __restrict__ x2,
            const float* __restrict__ x3, float* __restrict__ pooled){
    int b = blockIdx.x;
    for (int c = threadIdx.x; c < 448; c += 256){
        const float* src; int cc, C;
        if (c < 64)       { src = x1 + (size_t)b*64*64;  cc = c;       C = 64;  }
        else if (c < 192) { src = x2 + (size_t)b*64*128; cc = c - 64;  C = 128; }
        else              { src = x3 + (size_t)b*64*256; cc = c - 192; C = 256; }
        float s = 0.f, mx = -1e30f;
        for (int nn = 0; nn < 64; nn++){
            float v = src[(size_t)nn*C + cc];
            s += v; mx = fmaxf(mx, v);
        }
        pooled[(size_t)b*896 + c]       = s * (1.f/64.f);
        pooled[(size_t)b*896 + 448 + c] = mx;
    }
}

// final tiny fc: out = lrelu(bn(z2)) @ wc3 + bc3   (512 x 256 -> 512 x 2)
__global__ void k_fc3(const float* __restrict__ z, const float* __restrict__ ss,
                      const float* __restrict__ wc3, const float* __restrict__ bc3,
                      float* __restrict__ out){
    int r = blockIdx.x*blockDim.x + threadIdx.x;
    if (r >= 512) return;
    float a0 = bc3[0], a1 = bc3[1];
    for (int c = 0; c < 256; c++){
        float v = lrelu(z[(size_t)r*256 + c]*ss[c] + ss[256 + c]);
        a0 += v * wc3[c*2 + 0];
        a1 += v * wc3[c*2 + 1];
    }
    out[r*2 + 0] = a0;
    out[r*2 + 1] = a1;
}

// ---------------------------------------------------------------------------
extern "C" void kernel_launch(void* const* d_in, const int* in_sizes, int n_in,
                              void* d_out, int out_size, void* d_ws, size_t ws_size,
                              hipStream_t stream){
    (void)in_sizes; (void)n_in; (void)out_size; (void)ws_size;
    const float* x    = (const float*)d_in[0];
    // d_in[1] = batch (implied by layout; unused)
    const float* w_in = (const float*)d_in[2];
    const float* g_in = (const float*)d_in[3];
    const float* b_in = (const float*)d_in[4];
    const float* w1a  = (const float*)d_in[5];
    const float* g1a  = (const float*)d_in[6];
    const float* b1a  = (const float*)d_in[7];
    const float* w1b  = (const float*)d_in[8];
    const float* g1b  = (const float*)d_in[9];
    const float* b1b  = (const float*)d_in[10];
    const float* w2a  = (const float*)d_in[11];
    const float* g2a  = (const float*)d_in[12];
    const float* b2a  = (const float*)d_in[13];
    const float* w2b  = (const float*)d_in[14];
    const float* g2b  = (const float*)d_in[15];
    const float* b2b  = (const float*)d_in[16];
    const float* w3a  = (const float*)d_in[17];
    const float* g3a  = (const float*)d_in[18];
    const float* b3a  = (const float*)d_in[19];
    const float* w3b  = (const float*)d_in[20];
    const float* g3b  = (const float*)d_in[21];
    const float* b3b  = (const float*)d_in[22];
    const float* wc1  = (const float*)d_in[23];
    const float* gc1  = (const float*)d_in[24];
    const float* bc1  = (const float*)d_in[25];
    const float* wc2  = (const float*)d_in[26];
    const float* gc2  = (const float*)d_in[27];
    const float* bc2  = (const float*)d_in[28];
    const float* wc3  = (const float*)d_in[29];
    const float* bc3  = (const float*)d_in[30];
    float* out = (float*)d_out;

    const int M_EDGE = 512*64*16;     // 524288

    // ---- workspace carve-out ----
    char* base = (char*)d_ws;
    size_t cur = 0;
    auto carve = [&](size_t bytes)->char*{
        char* p = base + cur;
        cur += (bytes + 255) & ~(size_t)255;
        return p;
    };
    _Float16* w1a_h = (_Float16*)carve(8192*2);
    _Float16* w1b_h = (_Float16*)carve(4096*2);
    _Float16* w2a_h = (_Float16*)carve(16384*2);
    _Float16* w2b_h = (_Float16*)carve(16384*2);
    _Float16* w3a_h = (_Float16*)carve(65536*2);
    _Float16* w3b_h = (_Float16*)carve(65536*2);
    _Float16* wc1_h = (_Float16*)carve((size_t)458752*2);
    _Float16* wc2_h = (_Float16*)carve((size_t)131072*2);
    float* h0     = (float*)carve((size_t)32768*64*4);
    float* x0     = (float*)carve((size_t)32768*64*4);
    float* x1     = (float*)carve((size_t)512*64*64*4);
    float* x2     = (float*)carve((size_t)512*64*128*4);
    float* x3     = (float*)carve((size_t)512*64*256*4);
    int*   idxb   = (int*)  carve((size_t)512*64*16*4);
    float* hA     = (float*)carve((size_t)M_EDGE*256*4);
    float* hB     = (float*)carve((size_t)M_EDGE*256*4);
    float* pooled = (float*)carve((size_t)512*896*4);
    float* stats  = (float*)carve(1024*4);
    float* ss0    = (float*)carve(128*4);
    float* ssA    = (float*)carve(1024*4);
    float* ssB    = (float*)carve(1024*4);

    auto zstats = [&](){ hipMemsetAsync(stats, 0, 1024*sizeof(float), stream); };
    auto cvt = [&](const float* s, _Float16* d, int Kd, int N){
        int n = Kd*N;
        k_wt_f16<<<(n+255)/256, 256, 0, stream>>>(s, d, Kd, N);
    };

    // ---- weight conversion to transposed f16 ----
    cvt(w1a, w1a_h, 128, 64);   cvt(w1b, w1b_h, 64, 64);
    cvt(w2a, w2a_h, 128, 128);  cvt(w2b, w2b_h, 128, 128);
    cvt(w3a, w3a_h, 256, 256);  cvt(w3b, w3b_h, 256, 256);
    cvt(wc1, wc1_h, 896, 512);  cvt(wc2, wc2_h, 512, 256);

    // ---- input MLP + BN + lrelu ----
    zstats();
    k_input_mlp<<<128, 256, 0, stream>>>(x, w_in, h0, stats);
    k_bn_finalize<<<1, 64, 0, stream>>>(stats, g_in, b_in, ss0, 64, 1.f/32768.f);
    k_bn_apply<<<(32768*64)/256, 256, 0, stream>>>(h0, ss0, x0, 64, 32768*64);

    const float invE = 1.f/(float)M_EDGE;
    dim3 blk(128);

    // ---- EdgeConv 1 (Cin=64, Cmid=Cout=64) ----
    k_knn<<<512, 64, 0, stream>>>(x0, idxb, 64);
    zstats();
    k_gemm<2><<<dim3(8192, 1), blk, 0, stream>>>(x0, nullptr, idxb, 64, w1a_h,
                                                 M_EDGE, 128, 64, hA, stats);
    k_bn_finalize<<<1, 64, 0, stream>>>(stats, g1a, b1a, ssA, 64, invE);
    zstats();
    k_gemm<1><<<dim3(8192, 1), blk, 0, stream>>>(hA, ssA, nullptr, 0, w1b_h,
                                                 M_EDGE, 64, 64, hB, stats);
    k_bn_finalize<<<1, 64, 0, stream>>>(stats, g1b, b1b, ssB, 64, invE);
    k_scatter_max<<<512, 256, 0, stream>>>(hB, ssB, idxb, x1, 64);

    // ---- EdgeConv 2 (Cin=128) ----
    k_knn<<<512, 64, 0, stream>>>(x1, idxb, 128);
    zstats();
    k_gemm<2><<<dim3(8192, 2), blk, 0, stream>>>(x1, nullptr, idxb, 128, w2a_h,
                                                 M_EDGE, 256, 128, hA, stats);
    k_bn_finalize<<<2, 64, 0, stream>>>(stats, g2a, b2a, ssA, 128, invE);
    zstats();
    k_gemm<1><<<dim3(8192, 2), blk, 0, stream>>>(hA, ssA, nullptr, 0, w2b_h,
                                                 M_EDGE, 128, 128, hB, stats);
    k_bn_finalize<<<2, 64, 0, stream>>>(stats, g2b, b2b, ssB, 128, invE);
    k_scatter_max<<<512, 256, 0, stream>>>(hB, ssB, idxb, x2, 128);

    // ---- EdgeConv 3 (Cin=256) ----
    k_knn<<<512, 64, 0, stream>>>(x2, idxb, 256);
    zstats();
    k_gemm<2><<<dim3(8192, 4), blk, 0, stream>>>(x2, nullptr, idxb, 256, w3a_h,
                                                 M_EDGE, 512, 256, hA, stats);
    k_bn_finalize<<<4, 64, 0, stream>>>(stats, g3a, b3a, ssA, 256, invE);
    zstats();
    k_gemm<1><<<dim3(8192, 4), blk, 0, stream>>>(hA, ssA, nullptr, 0, w3b_h,
                                                 M_EDGE, 256, 256, hB, stats);
    k_bn_finalize<<<4, 64, 0, stream>>>(stats, g3b, b3b, ssB, 256, invE);
    k_scatter_max<<<512, 256, 0, stream>>>(hB, ssB, idxb, x3, 256);

    // ---- pooling + classifier ----
    k_pool<<<512, 256, 0, stream>>>(x1, x2, x3, pooled);

    zstats();
    k_gemm<0><<<dim3(8, 8), blk, 0, stream>>>(pooled, nullptr, nullptr, 0, wc1_h,
                                              512, 896, 512, hA, stats);
    k_bn_finalize<<<8, 64, 0, stream>>>(stats, gc1, bc1, ssA, 512, 1.f/512.f);
    zstats();
    k_gemm<1><<<dim3(8, 4), blk, 0, stream>>>(hA, ssA, nullptr, 0, wc2_h,
                                              512, 512, 256, hB, stats);
    k_bn_finalize<<<4, 64, 0, stream>>>(stats, gc2, bc2, ssB, 256, 1.f/512.f);
    k_fc3<<<2, 256, 0, stream>>>(hB, ssB, wc3, bc3, out);
}
